// MHCBlock_83425444757679
// MI455X (gfx1250) — matrix operand order repaired and checked by reading, not compiled, not run on hardware
//
#include <hip/hip_runtime.h>
#include <hip/hip_bf16.h>
#include <math.h>

typedef __attribute__((ext_vector_type(16))) _Float16 v16h;
typedef __attribute__((ext_vector_type(8)))  float    v8f;
typedef __attribute__((ext_vector_type(4)))  unsigned v4u;

union Frag {
  v16h h;
  _Float16 e[16];
  unsigned u[8];
};

#define EPS_RMS 1.1920929e-07f
#define QSCALE 0.17677669529663687f   // 1/sqrt(32)

__device__ __forceinline__ float sigmoidf_(float v) {
  return __builtin_amdgcn_rcpf(1.f + __expf(-v));
}
__device__ __forceinline__ float siluf_(float v) { return v * sigmoidf_(v); }

// load a B fragment: 8 contiguous dwords at p
__device__ __forceinline__ Frag ldfragB(const unsigned* p) {
  Frag f;
  *(v4u*)&f.u[0] = *(const v4u*)(p);
  *(v4u*)&f.u[4] = *(const v4u*)(p + 4);
  return f;
}
// load an A fragment: dwords [0..3] at p, [4..7] at p+8
__device__ __forceinline__ Frag ldfragA(const unsigned* p) {
  Frag f;
  *(v4u*)&f.u[0] = *(const v4u*)(p);
  *(v4u*)&f.u[4] = *(const v4u*)(p + 8);
  return f;
}

// ---------------------------------------------------------------------------
// Pack B (KxN f32) into WMMA fragment order:
// dword index = ((nt*kblocks)+kblk)*256 + lane*8 + v
// mode: 0=row-major, 1=engram-key permuted (4,256,128), 2=row-major with
//       cols<128 scaled by 1/sqrt(32) (fold attention scale into Wq)
// ---------------------------------------------------------------------------
__global__ void k_pack_b(const float* __restrict__ in, _Float16* __restrict__ out,
                         int N, int K, int mode)
{
  int idx = blockIdx.x * blockDim.x + threadIdx.x;   // one per output dword
  int total = (K * N) >> 1;
  if (idx >= total) return;
  int kblocks = K >> 5;
  int r = idx & 255;
  int tile = idx >> 8;
  int kblk = tile % kblocks;
  int nt = tile / kblocks;
  int lane = r >> 3, v = r & 7;
  int nl = lane & 15, kh = lane >> 4;
  int k0 = (kblk << 5) + ((v >> 2) << 4) + (kh << 3) + ((v & 3) << 1);
  int ncol = (nt << 4) + nl;
  float s0, s1;
  if (mode == 1) {  // key_w (4,256,128): col j -> s=j>>7, c=j&127; row d=k0
    int s = ncol >> 7, c = ncol & 127;
    s0 = in[(size_t)(s * 256 + k0) * 128 + c];
    s1 = in[(size_t)(s * 256 + k0 + 1) * 128 + c];
  } else {
    s0 = in[(size_t)k0 * N + ncol];
    s1 = in[(size_t)(k0 + 1) * N + ncol];
  }
  float scl = (mode == 2 && ncol < 128) ? QSCALE : 1.f;
  out[(size_t)idx * 2]     = (_Float16)(s0 * scl);
  out[(size_t)idx * 2 + 1] = (_Float16)(s1 * scl);
}

__global__ void k_qkvbias(const float* __restrict__ in, float* __restrict__ out)
{
  int i = blockIdx.x * blockDim.x + threadIdx.x;
  if (i < 384) out[i] = in[i] * (i < 128 ? QSCALE : 1.f);
}

// ---------------------------------------------------------------------------
// WMMA GEMM, software-pipelined with NAMED double-buffer fragments (no arrays,
// so nothing can be demoted to scratch). One wave computes 16x64.
// kblocks (K/32) must be even — true for K in {128,256,512}.
// ---------------------------------------------------------------------------
__global__ void k_gemm_f16(const _Float16* __restrict__ A,
                           const _Float16* __restrict__ Bp,
                           const float* __restrict__ bias,
                           float* __restrict__ of32,
                           _Float16* __restrict__ of16,
                           int M, int N, int K, int act)
{
  const int lane = threadIdx.x;
  const int m0 = blockIdx.x << 4;
  const int nt0 = blockIdx.y << 2;          // first of 4 N-tiles
  const int nl = lane & 15, kh = lane >> 4;
  const int arow = m0 + nl;
  const int kblocks = K >> 5;
  const unsigned* ap = (const unsigned*)A + (((size_t)arow * K) >> 1) + (kh << 2);
  const unsigned* bp0 = (const unsigned*)Bp + (size_t)nt0 * kblocks * 256 + (lane << 3);
  const size_t btile = (size_t)kblocks * 256;
  v8f acc0 = {}, acc1 = {}, acc2 = {}, acc3 = {};
  // preload kb=0 into X set
  Frag aX  = ldfragA(ap);
  Frag b0X = ldfragB(bp0);
  Frag b1X = ldfragB(bp0 + btile);
  Frag b2X = ldfragB(bp0 + 2 * btile);
  Frag b3X = ldfragB(bp0 + 3 * btile);
  for (int kb = 0; kb < kblocks; kb += 2) {
    // prefetch kb+1 into Y set
    const unsigned* bq = bp0 + (size_t)(kb + 1) * 256;
    Frag aY  = ldfragA(ap + (size_t)(kb + 1) * 16);
    Frag b0Y = ldfragB(bq);
    Frag b1Y = ldfragB(bq + btile);
    Frag b2Y = ldfragB(bq + 2 * btile);
    Frag b3Y = ldfragB(bq + 3 * btile);
    acc0 = __builtin_amdgcn_wmma_f32_16x16x32_f16(false, aX.h, false, b0X.h, (short)0, acc0, false, false);
    acc1 = __builtin_amdgcn_wmma_f32_16x16x32_f16(false, aX.h, false, b1X.h, (short)0, acc1, false, false);
    acc2 = __builtin_amdgcn_wmma_f32_16x16x32_f16(false, aX.h, false, b2X.h, (short)0, acc2, false, false);
    acc3 = __builtin_amdgcn_wmma_f32_16x16x32_f16(false, aX.h, false, b3X.h, (short)0, acc3, false, false);
    if (kb + 2 < kblocks) {   // prefetch kb+2 into X set
      const unsigned* br = bp0 + (size_t)(kb + 2) * 256;
      aX  = ldfragA(ap + (size_t)(kb + 2) * 16);
      b0X = ldfragB(br);
      b1X = ldfragB(br + btile);
      b2X = ldfragB(br + 2 * btile);
      b3X = ldfragB(br + 3 * btile);
    }
    acc0 = __builtin_amdgcn_wmma_f32_16x16x32_f16(false, aY.h, false, b0Y.h, (short)0, acc0, false, false);
    acc1 = __builtin_amdgcn_wmma_f32_16x16x32_f16(false, aY.h, false, b1Y.h, (short)0, acc1, false, false);
    acc2 = __builtin_amdgcn_wmma_f32_16x16x32_f16(false, aY.h, false, b2Y.h, (short)0, acc2, false, false);
    acc3 = __builtin_amdgcn_wmma_f32_16x16x32_f16(false, aY.h, false, b3Y.h, (short)0, acc3, false, false);
  }
  const v8f* accs[4] = {&acc0, &acc1, &acc2, &acc3};
#pragma unroll
  for (int ct = 0; ct < 4; ++ct) {
    int ncol = ((nt0 + ct) << 4) + nl;
    float bv = bias ? bias[ncol] : 0.f;
#pragma unroll
    for (int r = 0; r < 8; ++r) {
      int m = m0 + (kh << 3) + r;
      float v = (*accs[ct])[r] + bv;
      if (act == 1) v = siluf_(v);
      if (of32) of32[(size_t)m * N + ncol] = v;
      if (of16) of16[(size_t)m * N + ncol] = (_Float16)v;
    }
  }
}

// ---------------------------------------------------------------------------
// engram: ngram hash (+segment offsets baked in)
// ---------------------------------------------------------------------------
__global__ void k_hash(const int* __restrict__ ids, const int* __restrict__ mults,
                       int* __restrict__ hout, int BT)
{
  int bt = blockIdx.x * blockDim.x + threadIdx.x;
  if (bt >= BT) return;
  int t = bt & 127;
  unsigned id2 = (unsigned)ids[bt];
  unsigned id1 = (t >= 1) ? (unsigned)ids[bt - 1] : 0u;
  unsigned id0 = (t >= 2) ? (unsigned)ids[bt - 2] : 0u;
#pragma unroll
  for (int head = 0; head < 2; ++head) {
    unsigned m0 = (unsigned)mults[head * 3 + 0];
    unsigned m1 = (unsigned)mults[head * 3 + 1];
    unsigned h2 = (id1 * m0) ^ (id2 * m1);
    hout[bt * 4 + head] = (int)(h2 & 4095u) + head * 4096;
    unsigned n0 = (unsigned)mults[6 + head * 3 + 0];
    unsigned n1 = (unsigned)mults[6 + head * 3 + 1];
    unsigned n2 = (unsigned)mults[6 + head * 3 + 2];
    unsigned h3 = (id0 * n0) ^ (id1 * n1) ^ (id2 * n2);
    hout[bt * 4 + 2 + head] = (int)(h3 & 4095u) + (2 + head) * 4096;
  }
}

// gather embedding rows -> f16, two elements (one dword) per thread
__global__ void k_gather(const int* __restrict__ hids, const float* __restrict__ table,
                         _Float16* __restrict__ emb16, int total2)
{
  int idx = blockIdx.x * blockDim.x + threadIdx.x;
  if (idx >= total2) return;
  int d2 = idx & 127;               // dword within 256-wide row
  int bt = idx >> 7;
  int d = d2 << 1;
  int j = d >> 6, dd = d & 63;
  const float* src = table + (size_t)hids[bt * 4 + j] * 64 + dd;
  union { _Float16 h[2]; unsigned u; } pk;
  pk.h[0] = (_Float16)src[0];
  pk.h[1] = (_Float16)src[1];
  ((unsigned*)emb16)[idx] = pk.u;
}

// ---------------------------------------------------------------------------
// engram gating fused with conv-input rmsnorm. One 128-thread block per (bt,s).
// ---------------------------------------------------------------------------
__global__ void k_gate(const float* __restrict__ X, const float* __restrict__ kraw,
                       const float* __restrict__ vbase,
                       const float* __restrict__ nq_w, const float* __restrict__ nk_w,
                       const float* __restrict__ cn_w,
                       float* __restrict__ vg, float* __restrict__ xnorm)
{
  __shared__ float r1[128], r2[128];
  int bt = blockIdx.x >> 2;
  int s = blockIdx.x & 3;
  int tid = threadIdx.x;
  const float* xr = X + (size_t)bt * 512 + s * 128;
  const float* kr = kraw + (size_t)bt * 512 + s * 128;
  float xv = xr[tid], kv = kr[tid];
  r1[tid] = xv * xv; r2[tid] = kv * kv; __syncthreads();
  for (int off = 64; off > 0; off >>= 1) {
    if (tid < off) { r1[tid] += r1[tid + off]; r2[tid] += r2[tid + off]; }
    __syncthreads();
  }
  float qn = xv * rsqrtf(r1[0] * (1.f / 128.f) + EPS_RMS) * nq_w[s * 128 + tid];
  float kn = kv * rsqrtf(r2[0] * (1.f / 128.f) + EPS_RMS) * nk_w[s * 128 + tid];
  __syncthreads();
  r1[tid] = qn * kn; __syncthreads();
  for (int off = 64; off > 0; off >>= 1) {
    if (tid < off) r1[tid] += r1[tid + off];
    __syncthreads();
  }
  float gate = sigmoidf_(r1[0] * 0.08838834764831845f); // 1/sqrt(128)
  float vgv = vbase[(size_t)bt * 128 + tid] * gate;
  vg[(size_t)bt * 512 + s * 128 + tid] = vgv;
  __syncthreads();
  r1[tid] = vgv * vgv; __syncthreads();
  for (int off = 64; off > 0; off >>= 1) {
    if (tid < off) r1[tid] += r1[tid + off];
    __syncthreads();
  }
  xnorm[(size_t)bt * 512 + s * 128 + tid] =
      vgv * rsqrtf(r1[0] * (1.f / 128.f) + EPS_RMS) * cn_w[s * 128 + tid];
}

// depthwise causal conv + silu + residual, float4-vectorized
__global__ void k_conv_res(const float4* __restrict__ x0, const float4* __restrict__ vg,
                           const float4* __restrict__ xn, const float* __restrict__ cw,
                           float4* __restrict__ x1, int total4)
{
  int idx = blockIdx.x * blockDim.x + threadIdx.x;
  if (idx >= total4) return;
  int c4 = idx & 127;               // float4 within 512-wide row
  int bt = idx >> 7;
  int t = bt & 127;
  int ch0 = c4 << 2;
  float av[4] = {0.f, 0.f, 0.f, 0.f};
#pragma unroll
  for (int k = 0; k < 4; ++k) {
    int tt = t - 3 + k;
    if (tt < 0) continue;
    float4 v = xn[(size_t)(bt - 3 + k) * 128 + c4];
    av[0] += v.x * cw[(ch0 + 0) * 4 + k];
    av[1] += v.y * cw[(ch0 + 1) * 4 + k];
    av[2] += v.z * cw[(ch0 + 2) * 4 + k];
    av[3] += v.w * cw[(ch0 + 3) * 4 + k];
  }
  float4 xv = x0[idx], gv = vg[idx];
  float4 o;
  o.x = xv.x + gv.x + siluf_(av[0]);
  o.y = xv.y + gv.y + siluf_(av[1]);
  o.z = xv.z + gv.z + siluf_(av[2]);
  o.w = xv.w + gv.w + siluf_(av[3]);
  x1[idx] = o;
}

// ---------------------------------------------------------------------------
// MHC coefficients: per-token rmsnorm(512), 24 dots, sinkhorn(4x4, 20 it),
// x_agg + its rmsnorm -> f16. One 128-thread block per token.
// ---------------------------------------------------------------------------
__global__ void k_coef(const float* __restrict__ X,
                       const float* __restrict__ coef_w,
                       const float* __restrict__ pre_w,  const float* __restrict__ pre_b,
                       const float* __restrict__ post_w, const float* __restrict__ post_b,
                       const float* __restrict__ res_w,  const float* __restrict__ res_b,
                       const float* __restrict__ a_pre,  const float* __restrict__ a_post,
                       const float* __restrict__ a_res,
                       const float* __restrict__ ln_w,
                       _Float16* __restrict__ xa16,
                       float* __restrict__ Hres, float* __restrict__ Hpost_sum)
{
  __shared__ float part[24 * 128];
  __shared__ float red[128];
  __shared__ float Hpre[4];
  __shared__ float Hpost_v[4];
  __shared__ float logit[16];
  int bt = blockIdx.x;
  int tid = threadIdx.x;
  const float* xr = X + (size_t)bt * 512;
  float xv[4]; float ss = 0.f;
#pragma unroll
  for (int j = 0; j < 4; ++j) { float v = xr[tid + 128 * j]; xv[j] = v; ss += v * v; }
  red[tid] = ss; __syncthreads();
  for (int off = 64; off > 0; off >>= 1) {
    if (tid < off) red[tid] += red[tid + off];
    __syncthreads();
  }
  float inv = rsqrtf(red[0] * (1.f / 512.f) + EPS_RMS);
  float xn[4];
#pragma unroll
  for (int j = 0; j < 4; ++j) xn[j] = xv[j] * inv * coef_w[tid + 128 * j];
  for (int o = 0; o < 24; ++o) {
    float p = 0.f;
#pragma unroll
    for (int j = 0; j < 4; ++j) {
      int d = tid + 128 * j;
      float w = (o < 4) ? pre_w[d * 4 + o]
              : (o < 8) ? post_w[d * 4 + (o - 4)]
                        : res_w[d * 16 + (o - 8)];
      p += xn[j] * w;
    }
    part[o * 128 + tid] = p;
  }
  __syncthreads();
  for (int off = 64; off > 0; off >>= 1) {
    if (tid < off) {
      for (int o = 0; o < 24; ++o) part[o * 128 + tid] += part[o * 128 + tid + off];
    }
    __syncthreads();
  }
  if (tid < 4) {
    Hpre[tid] = sigmoidf_(a_pre[0] * (part[tid * 128] + pre_b[tid]));
  } else if (tid < 8) {
    Hpost_v[tid - 4] = 2.f * sigmoidf_(a_post[0] * (part[tid * 128] + post_b[tid - 4]));
  } else if (tid < 24) {
    logit[tid - 8] = a_res[0] * (part[tid * 128] + res_b[tid - 8]);
  }
  __syncthreads();
  if (tid == 0) {
    Hpost_sum[bt] = Hpost_v[0] + Hpost_v[1] + Hpost_v[2] + Hpost_v[3];
    float m[16];
    for (int i = 0; i < 4; ++i) {
      float l0 = logit[4*i], l1 = logit[4*i+1], l2 = logit[4*i+2], l3 = logit[4*i+3];
      float mx = fmaxf(fmaxf(l0, l1), fmaxf(l2, l3));
      m[4*i+0] = __expf(l0 - mx); m[4*i+1] = __expf(l1 - mx);
      m[4*i+2] = __expf(l2 - mx); m[4*i+3] = __expf(l3 - mx);
    }
    for (int it = 0; it < 20; ++it) {
      for (int i = 0; i < 4; ++i) {
        float s = m[4*i] + m[4*i+1] + m[4*i+2] + m[4*i+3] + 1e-6f;
        float r1 = 1.f / s;
        m[4*i] *= r1; m[4*i+1] *= r1; m[4*i+2] *= r1; m[4*i+3] *= r1;
      }
      for (int j = 0; j < 4; ++j) {
        float s = m[j] + m[4+j] + m[8+j] + m[12+j] + 1e-6f;
        float r1 = 1.f / s;
        m[j] *= r1; m[4+j] *= r1; m[8+j] *= r1; m[12+j] *= r1;
      }
    }
    for (int i = 0; i < 16; ++i) Hres[(size_t)bt * 16 + i] = m[i];
  }
  __syncthreads();
  float a0 = Hpre[0] * xr[tid] + Hpre[1] * xr[128 + tid] +
             Hpre[2] * xr[256 + tid] + Hpre[3] * xr[384 + tid];
  red[tid] = a0 * a0; __syncthreads();
  for (int off = 64; off > 0; off >>= 1) {
    if (tid < off) red[tid] += red[tid + off];
    __syncthreads();
  }
  float inv2 = rsqrtf(red[0] * (1.f / 128.f) + EPS_RMS);
  xa16[(size_t)bt * 128 + tid] = (_Float16)(a0 * inv2 * ln_w[tid]);
}

// ---------------------------------------------------------------------------
// Causal attention, one block per (b, h). Reads f16 QKV (scale pre-folded
// into Wq). All WMMA fragments load as 2x ds b128 (V transposed, stride 136).
// ---------------------------------------------------------------------------
#define VSTRIDE 136
__global__ void k_attn(const _Float16* __restrict__ qkv16, _Float16* __restrict__ y16)
{
  const int T = 128, D = 32;
  int b = blockIdx.x >> 2;
  int h = blockIdx.x & 3;
  extern __shared__ char smem[];
  _Float16* Qs = (_Float16*)smem;                       // 4096 halfs
  _Float16* Ks = Qs + 4096;                             // 4096 halfs
  _Float16* Vt = Ks + 4096;                             // 32*136 = 4352 halfs
  float*    Ss = (float*)(smem + (4096 + 4096 + 4352) * 2);   // 16384 f32
  _Float16* Ps = (_Float16*)((char*)Ss + 16384 * 4);          // 16384 halfs
  int tid = threadIdx.x;
  const float NEG_INF = -__builtin_inff();
  const _Float16* base = qkv16 + (size_t)b * T * 384 + h * 32;
  for (int i = tid; i < 2048; i += 256) {               // Q,K: dword copies
    int t = i >> 4, dp = i & 15;
    ((unsigned*)Qs)[t * 16 + dp] = *((const unsigned*)(base + (size_t)t * 384) + dp);
    ((unsigned*)Ks)[t * 16 + dp] = *((const unsigned*)(base + (size_t)t * 384 + 128) + dp);
  }
  for (int i = tid; i < 4096; i += 256) {               // V transpose
    int t = i >> 5, d = i & 31;
    Vt[d * VSTRIDE + t] = base[(size_t)t * 384 + 256 + d];
  }
  __syncthreads();
  int lane = tid & 31;
  int nl = lane & 15, kh = lane >> 4;
  int m0 = (tid >> 5) << 4;
  // scores: S = (Q*scale) K^T, causal mask
  Frag aq;
  *(v4u*)&aq.u[0] = *(const v4u*)(Qs + (m0 + nl) * D + (kh << 4));
  *(v4u*)&aq.u[4] = *(const v4u*)(Qs + (m0 + nl) * D + (kh << 4) + 8);
  for (int cn = 0; cn < 8; ++cn) {
    Frag bf;
    *(v4u*)&bf.u[0] = *(const v4u*)(Ks + (cn * 16 + nl) * D + (kh << 4));
    *(v4u*)&bf.u[4] = *(const v4u*)(Ks + (cn * 16 + nl) * D + (kh << 4) + 8);
    v8f c = {};
    c = __builtin_amdgcn_wmma_f32_16x16x32_f16(false, aq.h, false, bf.h,
                                               (short)0, c, false, false);
#pragma unroll
    for (int r = 0; r < 8; ++r) {
      int m = m0 + (kh << 3) + r;
      int n = cn * 16 + nl;
      Ss[m * T + n] = (n <= m) ? c[r] : NEG_INF;
    }
  }
  __syncthreads();
  if (tid < T) {                                        // softmax, one row/thread
    float mx = NEG_INF;
    for (int j = 0; j < T; ++j) mx = fmaxf(mx, Ss[tid * T + j]);
    float s = 0.f;
    for (int j = 0; j < T; ++j) {
      float e = __expf(Ss[tid * T + j] - mx);
      Ss[tid * T + j] = e;
      s += e;
    }
    float invs = __builtin_amdgcn_rcpf(s);
    for (int j = 0; j < T; ++j)
      Ps[tid * T + j] = (_Float16)(Ss[tid * T + j] * invs);
  }
  __syncthreads();
  v8f acc0 = {}, acc1 = {};                             // Y = P @ V
  for (int kb = 0; kb < T; kb += 32) {
    Frag a;
    *(v4u*)&a.u[0] = *(const v4u*)(Ps + (m0 + nl) * T + kb + (kh << 4));
    *(v4u*)&a.u[4] = *(const v4u*)(Ps + (m0 + nl) * T + kb + (kh << 4) + 8);
    Frag b0, b1;
    *(v4u*)&b0.u[0] = *(const v4u*)(Vt + nl * VSTRIDE + kb + (kh << 4));
    *(v4u*)&b0.u[4] = *(const v4u*)(Vt + nl * VSTRIDE + kb + (kh << 4) + 8);
    *(v4u*)&b1.u[0] = *(const v4u*)(Vt + (16 + nl) * VSTRIDE + kb + (kh << 4));
    *(v4u*)&b1.u[4] = *(const v4u*)(Vt + (16 + nl) * VSTRIDE + kb + (kh << 4) + 8);
    acc0 = __builtin_amdgcn_wmma_f32_16x16x32_f16(false, a.h, false, b0.h,
                                                  (short)0, acc0, false, false);
    acc1 = __builtin_amdgcn_wmma_f32_16x16x32_f16(false, a.h, false, b1.h,
                                                  (short)0, acc1, false, false);
  }
#pragma unroll
  for (int r = 0; r < 8; ++r) {
    int m = m0 + (kh << 3) + r;
    y16[(size_t)(b * T + m) * 128 + h * D + nl]      = (_Float16)acc0[r];
    y16[(size_t)(b * T + m) * 128 + h * D + 16 + nl] = (_Float16)acc1[r];
  }
}

// ---------------------------------------------------------------------------
// MHC combine, float4-vectorized:
// out[bt,n,c] = sum_m Hres[bt,n,m]*xin[bt,m,c] + Hpost[bt]*y[bt,c]
// ---------------------------------------------------------------------------
__global__ void k_combine(const float4* __restrict__ xin, const float* __restrict__ Hres,
                          const float* __restrict__ Hpost, const float4* __restrict__ y,
                          float4* __restrict__ out, int total4)
{
  int idx = blockIdx.x * blockDim.x + threadIdx.x;
  if (idx >= total4) return;
  int c4 = idx & 31;                // float4 within 128-wide stream
  int nidx = (idx >> 5) & 3;
  int bt = idx >> 7;
  const float* hr = Hres + (size_t)bt * 16 + nidx * 4;
  const float4* xp = xin + (size_t)bt * 128;
  float4 v0 = xp[c4], v1 = xp[32 + c4], v2 = xp[64 + c4], v3 = xp[96 + c4];
  float4 yv = y[(size_t)bt * 32 + c4];
  float hp = Hpost[bt];
  float4 o;
  o.x = hr[0] * v0.x + hr[1] * v1.x + hr[2] * v2.x + hr[3] * v3.x + hp * yv.x;
  o.y = hr[0] * v0.y + hr[1] * v1.y + hr[2] * v2.y + hr[3] * v3.y + hp * yv.y;
  o.z = hr[0] * v0.z + hr[1] * v1.z + hr[2] * v2.z + hr[3] * v3.z + hp * yv.z;
  o.w = hr[0] * v0.w + hr[1] * v1.w + hr[2] * v2.w + hr[3] * v3.w + hp * yv.w;
  out[idx] = o;
}

// ---------------------------------------------------------------------------
extern "C" void kernel_launch(void* const* d_in, const int* in_sizes, int n_in,
                              void* d_out, int out_size, void* d_ws, size_t ws_size,
                              hipStream_t stream)
{
  (void)in_sizes; (void)n_in; (void)out_size; (void)ws_size;
  const float* x          = (const float*)d_in[0];
  const int*   ids        = (const int*)d_in[1];
  const int*   mults      = (const int*)d_in[2];
  const float* emb_table  = (const float*)d_in[3];
  const float* val_w      = (const float*)d_in[4];
  const float* val_b      = (const float*)d_in[5];
  const float* key_w      = (const float*)d_in[6];
  const float* key_b      = (const float*)d_in[7];
  const float* nq_w       = (const float*)d_in[8];
  const float* nk_w       = (const float*)d_in[9];
  const float* conv_w     = (const float*)d_in[10];
  const float* convnorm_w = (const float*)d_in[11];
  const float* a_coef_w   = (const float*)d_in[12];
  const float* a_pre_w    = (const float*)d_in[13];
  const float* a_pre_b    = (const float*)d_in[14];
  const float* a_post_w   = (const float*)d_in[15];
  const float* a_post_b   = (const float*)d_in[16];
  const float* a_res_w    = (const float*)d_in[17];
  const float* a_res_b    = (const float*)d_in[18];
  const float* a_al_pre   = (const float*)d_in[19];
  const float* a_al_post  = (const float*)d_in[20];
  const float* a_al_res   = (const float*)d_in[21];
  const float* a_ln_w     = (const float*)d_in[22];
  const float* qkv_w      = (const float*)d_in[23];
  const float* qkv_b      = (const float*)d_in[24];
  const float* proj_w     = (const float*)d_in[25];
  const float* proj_b     = (const float*)d_in[26];
  const float* m_coef_w   = (const float*)d_in[27];
  const float* m_pre_w    = (const float*)d_in[28];
  const float* m_pre_b    = (const float*)d_in[29];
  const float* m_post_w   = (const float*)d_in[30];
  const float* m_post_b   = (const float*)d_in[31];
  const float* m_res_w    = (const float*)d_in[32];
  const float* m_res_b    = (const float*)d_in[33];
  const float* m_al_pre   = (const float*)d_in[34];
  const float* m_al_post  = (const float*)d_in[35];
  const float* m_al_res   = (const float*)d_in[36];
  const float* m_ln_w     = (const float*)d_in[37];
  const float* mlp_w1     = (const float*)d_in[38];
  const float* mlp_b1     = (const float*)d_in[39];
  const float* mlp_w2     = (const float*)d_in[40];
  const float* mlp_b2     = (const float*)d_in[41];

  const int BT = 32768;
  const size_t MB = 1ull << 20;
  char* W = (char*)d_ws;
  // small region (< 16 MB)
  _Float16* wval16  = (_Float16*)(W + 0 * MB);
  _Float16* wkey16  = (_Float16*)(W + 1 * MB);
  _Float16* wqkv16  = (_Float16*)(W + 2 * MB);
  _Float16* wproj16 = (_Float16*)(W + 3 * MB);
  _Float16* wm1_16  = (_Float16*)(W + 4 * MB);
  _Float16* wm2_16  = (_Float16*)(W + 5 * MB);
  int*      hids    = (int*)(W + 6 * MB);
  float*    HresA   = (float*)(W + 7 * MB);
  float*    HpostA  = (float*)(W + 9 * MB);
  float*    HresM   = (float*)(W + 10 * MB);
  float*    HpostM  = (float*)(W + 12 * MB);
  float*    qkvb_s  = (float*)(W + 13 * MB);    // scaled qkv bias (384)
  // big overlaid buffers
  float*    bufA    = (float*)(W + 16 * MB);    // kraw(64M) -> qkv16(24M) -> h16(32M)
  float*    bufB    = (float*)(W + 80 * MB);    // vg(64M) -> x2(64M)
  float*    bufC    = (float*)(W + 144 * MB);   // xn(64M) -> yproj/y2/y16
  float*    bufD    = (float*)(W + 208 * MB);   // x1(64M)
  _Float16* emb16   = (_Float16*)(W + 272 * MB);
  float*    vbase   = (float*)(W + 288 * MB);
  _Float16* xa16    = (_Float16*)(W + 304 * MB);
  _Float16* xa2_16  = (_Float16*)(W + 312 * MB);
  float*    yproj   = bufC;
  float*    y2f     = (float*)((char*)bufC + 16 * MB);
  _Float16* y16     = (_Float16*)((char*)bufC + 48 * MB);
  _Float16* qkv16   = (_Float16*)bufA;
  _Float16* h16     = (_Float16*)bufA;

  // --- weight packing (f32 -> WMMA-fragment-ordered f16) ---
  k_pack_b<<<(256 * 128 / 2 + 255) / 256, 256, 0, stream>>>(val_w, wval16, 128, 256, 0);
  k_pack_b<<<(256 * 512 / 2 + 255) / 256, 256, 0, stream>>>(key_w, wkey16, 512, 256, 1);
  k_pack_b<<<(128 * 384 / 2 + 255) / 256, 256, 0, stream>>>(qkv_w, wqkv16, 384, 128, 2);
  k_pack_b<<<(128 * 128 / 2 + 255) / 256, 256, 0, stream>>>(proj_w, wproj16, 128, 128, 0);
  k_pack_b<<<(128 * 512 / 2 + 255) / 256, 256, 0, stream>>>(mlp_w1, wm1_16, 512, 128, 0);
  k_pack_b<<<(512 * 128 / 2 + 255) / 256, 256, 0, stream>>>(mlp_w2, wm2_16, 128, 512, 0);
  k_qkvbias<<<2, 256, 0, stream>>>(qkv_b, qkvb_s);

  // --- engram ---
  k_hash<<<BT / 256, 256, 0, stream>>>(ids, mults, hids, BT);
  k_gather<<<(BT * 128) / 256, 256, 0, stream>>>(hids, emb_table, emb16, BT * 128);
  k_gemm_f16<<<dim3(BT / 16, 128 / 64), 32, 0, stream>>>(
      emb16, wval16, val_b, vbase, nullptr, BT, 128, 256, 0);
  k_gemm_f16<<<dim3(BT / 16, 512 / 64), 32, 0, stream>>>(
      emb16, wkey16, key_b, bufA, nullptr, BT, 512, 256, 0);
  k_gate<<<BT * 4, 128, 0, stream>>>(x, bufA, vbase, nq_w, nk_w, convnorm_w, bufB, bufC);
  k_conv_res<<<(BT * 128) / 256, 256, 0, stream>>>(
      (const float4*)x, (const float4*)bufB, (const float4*)bufC, conv_w,
      (float4*)bufD, BT * 128);

  // --- MHC(attention) ---
  k_coef<<<BT, 128, 0, stream>>>(bufD, a_coef_w, a_pre_w, a_pre_b, a_post_w, a_post_b,
                                 a_res_w, a_res_b, a_al_pre, a_al_post, a_al_res,
                                 a_ln_w, xa16, HresA, HpostA);
  k_gemm_f16<<<dim3(BT / 16, 384 / 64), 32, 0, stream>>>(
      xa16, wqkv16, qkvb_s, nullptr, qkv16, BT, 384, 128, 0);
  k_attn<<<1024, 256, (4096 + 4096 + 4352) * 2 + 16384 * 4 + 16384 * 2, stream>>>(qkv16, y16);
  k_gemm_f16<<<dim3(BT / 16, 128 / 64), 32, 0, stream>>>(
      y16, wproj16, proj_b, yproj, nullptr, BT, 128, 128, 0);
  k_combine<<<(BT * 128) / 256, 256, 0, stream>>>(
      (const float4*)bufD, HresA, HpostA, (const float4*)yproj, (float4*)bufB, BT * 128);

  // --- MHC(MLP) ---
  k_coef<<<BT, 128, 0, stream>>>(bufB, m_coef_w, m_pre_w, m_pre_b, m_post_w, m_post_b,
                                 m_res_w, m_res_b, m_al_pre, m_al_post, m_al_res,
                                 m_ln_w, xa2_16, HresM, HpostM);
  k_gemm_f16<<<dim3(BT / 16, 512 / 64), 32, 0, stream>>>(
      xa2_16, wm1_16, mlp_b1, nullptr, h16, BT, 512, 128, 1);
  k_gemm_f16<<<dim3(BT / 16, 128 / 64), 32, 0, stream>>>(
      h16, wm2_16, mlp_b2, y2f, nullptr, BT, 128, 512, 0);
  k_combine<<<(BT * 128) / 256, 256, 0, stream>>>(
      (const float4*)bufB, HresM, HpostM, (const float4*)y2f, (float4*)d_out, BT * 128);
}
